// WindowBasedSelfAttention_88347477279451
// MI455X (gfx1250) — matrix operand order repaired and checked by reading, not compiled, not run on hardware
//
#include <hip/hip_runtime.h>

typedef __attribute__((ext_vector_type(16))) _Float16 v16h;
typedef __attribute__((ext_vector_type(8)))  float    v8f;
typedef __attribute__((ext_vector_type(4)))  unsigned int v4u;
typedef __attribute__((ext_vector_type(8)))  int      v8i;
typedef __attribute__((ext_vector_type(4)))  int      v4i;

#if __has_builtin(__builtin_amdgcn_tensor_load_to_lds) && \
    __has_builtin(__builtin_amdgcn_s_wait_tensorcnt)
#define USE_TDM 1
#else
#define USE_TDM 0
#endif

namespace {

constexpr int kB  = 4;
constexpr int kS  = 2048;
constexpr int kD  = 1024;
constexpr int kDH = 64;
constexpr int kRows = kB * kS;                 // 8192 rows of X / Q / K / V
constexpr float kInvScale = 1.0f / 32.0f;      // 1/sqrt(d_model)

// A-operand element mapping for v_wmma_*_16x16x32 (ISA 7.12.2, 16-bit A 16x32)
__device__ __forceinline__ int amap(int j, int h) { return (j < 8 ? j : j + 8) + (h << 3); }
// B-operand mapping (same k order as A: K = 8h+j for j<8, 16+8h+(j-8) for j>=8)
__device__ __forceinline__ int bmap(int j, int h) { return (j < 8 ? j : j + 8) + (h << 3); }

__device__ __forceinline__ v8f wmma16(v16h a, v16h b, v8f c) {
    return __builtin_amdgcn_wmma_f32_16x16x32_f16(false, a, false, b, (short)0, c,
                                                  false, false);
}

#if USE_TDM
// Issue a TDM load of a 2-D f32 tile (tile_w x tile_h, row stride in elements)
// into LDS at byte offset lds_off.  D# layout per CDNA5 ISA.
// This toolchain exposes the 6-arg builtin:
//   (uint32x4 g0, int32x8 g1, int32x4 g2, int32x4 g3, int32x8 extra, i32 cpol)
__device__ __forceinline__ void tdm_load_tile_f32(const float* gptr, unsigned lds_off,
                                                  unsigned tile_w, unsigned tile_h,
                                                  unsigned tensor_w, unsigned tensor_h,
                                                  unsigned row_stride) {
    unsigned long long ga = (unsigned long long)(uintptr_t)gptr;
    v4u g0;
    g0[0] = 1u;                                        // count=1 (valid), user mode
    g0[1] = lds_off;                                   // lds_addr (bytes)
    g0[2] = (unsigned)(ga & 0xffffffffu);              // global_addr[31:0]
    g0[3] = (unsigned)((ga >> 32) & 0x01ffffffu)       // global_addr[56:32]
          | 0x80000000u;                               // type=2 ("image")
    v8i g1;
    g1[0] = (int)(2u << 16);                           // data_size=2 (4 bytes)
    g1[1] = (int)(tensor_w << 16);                     // tensor_dim0[15:0] @ [63:48]
    g1[2] = (int)((tensor_w >> 16) | (tensor_h << 16));// dim0[31:16] | dim1[15:0]
    g1[3] = (int)((tensor_h >> 16) | (tile_w << 16));  // dim1[31:16] | tile_dim0
    g1[4] = (int)tile_h;                               // tile_dim1 (tile_dim2=0)
    g1[5] = (int)row_stride;                           // tensor_dim0_stride[31:0]
    g1[6] = 0;                                         // stride hi / dim1_stride
    g1[7] = 0;
    v4i z4 = {0, 0, 0, 0};                             // groups 2/3 unused (2-D)
    v8i z8 = {0, 0, 0, 0, 0, 0, 0, 0};
    __builtin_amdgcn_tensor_load_to_lds(g0, g1, z4, z4, z8, 0);
}
#endif

// ---------------------------------------------------------------------------
// Kernel 1: Q/K/V projections.  X[8192,1024](f32) @ W[1024,64](f32) + b -> f16
// One wave computes a 16x64 output slab (4 n-tiles), K-loop in steps of 32.
// Grid: 3 matrices * 512 m-tiles = 1536 waves = 384 blocks of 128 threads.
// ---------------------------------------------------------------------------
__global__ void proj_kernel(const float* __restrict__ x,
                            const float* __restrict__ Wq, const float* __restrict__ bq,
                            const float* __restrict__ Wk, const float* __restrict__ bk,
                            const float* __restrict__ Wv, const float* __restrict__ bv,
                            _Float16* __restrict__ Qh, _Float16* __restrict__ Kh,
                            _Float16* __restrict__ Vh)
{
    const int wid  = threadIdx.x >> 5;
    const int lane = threadIdx.x & 31;
    const int wg   = blockIdx.x * 4 + wid;     // 0..1535
    const int w    = wg / 512;                 // which matrix
    const int mt   = wg % 512;                 // m-tile
    const int row0 = mt * 16;
    const int h    = lane >> 4;
    const int rc   = lane & 15;

    const float* Wm = (w == 0) ? Wq : (w == 1) ? Wk : Wv;
    const float* bm = (w == 0) ? bq : (w == 1) ? bk : bv;
    _Float16*   Out = (w == 0) ? Qh : (w == 1) ? Kh : Vh;

    v8f acc[4] = {};
    for (int k0 = 0; k0 < kD; k0 += 32) {
        v16h a;
        const float* xr = x + (size_t)(row0 + rc) * kD + k0;
        #pragma unroll
        for (int j = 0; j < 16; ++j) a[j] = (_Float16)xr[amap(j, h)];
        #pragma unroll
        for (int t = 0; t < 4; ++t) {
            v16h bf;
            #pragma unroll
            for (int j = 0; j < 16; ++j)
                bf[j] = (_Float16)Wm[(size_t)(k0 + bmap(j, h)) * kDH + t * 16 + rc];
            acc[t] = wmma16(a, bf, acc[t]);
        }
    }
    #pragma unroll
    for (int t = 0; t < 4; ++t) {
        const float bias = bm[t * 16 + rc];
        #pragma unroll
        for (int r = 0; r < 8; ++r) {
            const int m = r + 8 * h;
            Out[(size_t)(row0 + m) * kDH + t * 16 + rc] = (_Float16)(acc[t][r] + bias);
        }
    }
}

// ---------------------------------------------------------------------------
// Kernel 2: column softmax stats.  For each (b, 16 keys): linv = 1/sum_q exp(s/32).
// Scores have ~0.25 std here, so max-subtraction is unnecessary for safety.
// Grid: 4*128 = 512 waves = 128 blocks of 128 threads.
// ---------------------------------------------------------------------------
__global__ void colstats_kernel(const _Float16* __restrict__ Qh,
                                const _Float16* __restrict__ Kh,
                                float* __restrict__ linv)
{
    const int wid  = threadIdx.x >> 5;
    const int lane = threadIdx.x & 31;
    const int wg   = blockIdx.x * 4 + wid;     // 0..511
    const int b    = wg / 128;
    const int k0   = (wg % 128) * 16;
    const int h    = lane >> 4;
    const int rc   = lane & 15;

    v16h kb0, kb1;
    const _Float16* kr = Kh + (size_t)(b * kS + k0 + rc) * kDH;
    #pragma unroll
    for (int j = 0; j < 16; ++j) {
        kb0[j] = kr[bmap(j, h)];
        kb1[j] = kr[32 + bmap(j, h)];
    }

    float sum = 0.0f;
    for (int q0 = 0; q0 < kS; q0 += 16) {
        v16h a0, a1;
        const _Float16* qr = Qh + (size_t)(b * kS + q0 + rc) * kDH;
        #pragma unroll
        for (int j = 0; j < 16; ++j) {
            a0[j] = qr[amap(j, h)];
            a1[j] = qr[32 + amap(j, h)];
        }
        v8f c = {};
        c = wmma16(a0, kb0, c);
        c = wmma16(a1, kb1, c);
        #pragma unroll
        for (int r = 0; r < 8; ++r) sum += __expf(c[r] * kInvScale);
    }
    sum += __shfl_xor(sum, 16, 32);            // combine the two lane halves
    if (lane < 16) linv[b * kS + k0 + lane] = 1.0f / sum;
}

// ---------------------------------------------------------------------------
// Kernel 3: fused attention output.  One wave per workgroup (so all TDM
// descriptor fields derive from blockIdx -> provably wave-uniform SGPRs).
// Per (b, 16-query tile): stream mask tiles via TDM double-buffer, rebuild
// score tile with WMMA, P = exp(s/32)*linv[k] + mask, stage P as f16 in LDS,
// re-gather in A layout, WMMA-accumulate P@V.  Grid: 512 blocks of 32.
// ---------------------------------------------------------------------------
__global__ __launch_bounds__(32) void attn_kernel(const _Float16* __restrict__ Qh,
                            const _Float16* __restrict__ Kh,
                            const _Float16* __restrict__ Vh,
                            const float* __restrict__ linv,
                            const float* __restrict__ mask,
                            float* __restrict__ out)
{
    __shared__ _Float16 Pbuf[16 * 32];         // 16q x 32k P tile (f16)
    __shared__ float    Mbuf[2][16 * 32];      // double-buffered mask tiles

    const int lane = threadIdx.x;              // blockDim.x == 32
    const int b    = blockIdx.x >> 7;
    const int q0   = (blockIdx.x & 127) * 16;
    const int h    = lane >> 4;
    const int rc   = lane & 15;

    // Q tile (A operand) is fixed for this wave
    v16h a0, a1;
    const _Float16* qr = Qh + (size_t)(b * kS + q0 + rc) * kDH;
    #pragma unroll
    for (int j = 0; j < 16; ++j) {
        a0[j] = qr[amap(j, h)];
        a1[j] = qr[32 + amap(j, h)];
    }

    const float* mtile = mask + (size_t)(b * kS + q0) * kS;   // 16 x 2048 slab
    v8f acc[4] = {};

#if USE_TDM
    tdm_load_tile_f32(mtile, (unsigned)(uintptr_t)&Mbuf[0][0],
                      /*tile*/32, 16, /*tensor*/kS, 16, /*stride*/kS);
#endif
    int cur = 0;

    for (int k0 = 0; k0 < kS; k0 += 32) {
#if USE_TDM
        if (k0 + 32 < kS) {
            tdm_load_tile_f32(mtile + (k0 + 32),
                              (unsigned)(uintptr_t)&Mbuf[cur ^ 1][0],
                              32, 16, kS, 16, kS);
            __builtin_amdgcn_s_wait_tensorcnt(1);   // previous tile has landed
        } else {
            __builtin_amdgcn_s_wait_tensorcnt(0);
        }
        // Launder the LDS index so the compiler cannot fold reads of
        // TDM-written (to it: never-stored) shared memory to undef.
        unsigned mbase = (unsigned)(cur * (16 * 32));
        asm volatile("" : "+v"(mbase) : : "memory");
        const float* mrow = &Mbuf[0][0] + mbase;    // tile: [16 rows][32 cols]
#endif

        #pragma unroll
        for (int s = 0; s < 2; ++s) {
            const int kk = k0 + s * 16;
            v16h kb0, kb1;
            const _Float16* kr = Kh + (size_t)(b * kS + kk + rc) * kDH;
            #pragma unroll
            for (int j = 0; j < 16; ++j) {
                kb0[j] = kr[bmap(j, h)];
                kb1[j] = kr[32 + bmap(j, h)];
            }
            v8f sc = {};
            sc = wmma16(a0, kb0, sc);
            sc = wmma16(a1, kb1, sc);

            const float lv = linv[b * kS + kk + rc];   // column n = key kk+rc
            #pragma unroll
            for (int r = 0; r < 8; ++r) {
                const int m = r + 8 * h;               // query row in tile
#if USE_TDM
                const float mv = mrow[m * 32 + s * 16 + rc];
#else
                const float mv = mtile[(size_t)m * kS + kk + rc];
#endif
                const float p = __expf(sc[r] * kInvScale) * lv + mv;
                Pbuf[m * 32 + s * 16 + rc] = (_Float16)p;
            }
        }

        // Re-gather P in A layout (LDS ops are in-order within a wave)
        v16h pa;
        #pragma unroll
        for (int j = 0; j < 16; ++j) pa[j] = Pbuf[rc * 32 + amap(j, h)];

        #pragma unroll
        for (int t = 0; t < 4; ++t) {
            v16h vb;
            #pragma unroll
            for (int j = 0; j < 16; ++j)
                vb[j] = Vh[(size_t)(b * kS + k0 + bmap(j, h)) * kDH + t * 16 + rc];
            acc[t] = wmma16(pa, vb, acc[t]);
        }
        cur ^= 1;
    }

    #pragma unroll
    for (int t = 0; t < 4; ++t)
        #pragma unroll
        for (int r = 0; r < 8; ++r)
            out[(size_t)(b * kS + q0 + r + 8 * h) * kDH + t * 16 + rc] = acc[t][r];
}

} // namespace

extern "C" void kernel_launch(void* const* d_in, const int* in_sizes, int n_in,
                              void* d_out, int out_size, void* d_ws, size_t ws_size,
                              hipStream_t stream) {
    const float* x    = (const float*)d_in[0];
    const float* mask = (const float*)d_in[1];
    const float* Wq   = (const float*)d_in[2];
    const float* bq   = (const float*)d_in[3];
    const float* Wk   = (const float*)d_in[4];
    const float* bk   = (const float*)d_in[5];
    const float* Wv   = (const float*)d_in[6];
    const float* bv   = (const float*)d_in[7];
    float* out        = (float*)d_out;

    char* ws = (char*)d_ws;
    const size_t mat_bytes = (size_t)kRows * kDH * sizeof(_Float16);  // 1 MB each
    _Float16* Qh = (_Float16*)(ws);
    _Float16* Kh = (_Float16*)(ws + mat_bytes);
    _Float16* Vh = (_Float16*)(ws + 2 * mat_bytes);
    float*  linv = (float*)  (ws + 3 * mat_bytes);

    proj_kernel<<<dim3(384), dim3(128), 0, stream>>>(x, Wq, bq, Wk, bk, Wv, bv,
                                                     Qh, Kh, Vh);
    colstats_kernel<<<dim3(128), dim3(128), 0, stream>>>(Qh, Kh, linv);
    attn_kernel<<<dim3(512), dim3(32), 0, stream>>>(Qh, Kh, Vh, linv, mask, out);
}
